// DE_1082331759091
// MI455X (gfx1250) — matrix operand order repaired and checked by reading, not compiled, not run on hardware
//
#include <hip/hip_runtime.h>

// GIN forward: out[b] = colmax( relu( S*(relu(S*X*W1+b1))*W2 + b2 ) ), S = I + triu(adj,1)^T
// BF16 WMMA (16x16x32) with f32 accumulation; one workgroup (8 wave32) per graph.

#define N_NODES 128
#define D_IN    256
#define D_HID   512
#define D_OUT   256

typedef __attribute__((ext_vector_type(8)))  float  v8f;
typedef __attribute__((ext_vector_type(4)))  float  v4f;
typedef __attribute__((ext_vector_type(2)))  float  v2f;
typedef __attribute__((ext_vector_type(4)))  int    v4i;
typedef __attribute__((ext_vector_type(2)))  __bf16 v2bf;
typedef __attribute__((ext_vector_type(4)))  __bf16 v4bf;
typedef __attribute__((ext_vector_type(8)))  __bf16 v8bf;
typedef __attribute__((ext_vector_type(16))) __bf16 v16bf;

__device__ __forceinline__ v16bf cat16(v8bf lo, v8bf hi) {
  return __builtin_shufflevector(lo, hi, 0,1,2,3,4,5,6,7,8,9,10,11,12,13,14,15);
}

// A-fragment (16x32 bf16) from a row-major buffer (pitch in halves, pitch%8==0).
// lane L: row M = L&15, K = kb..kb+7 and 16+kb..16+kb+7, kb = (L>=16)*8.
__device__ __forceinline__ v16bf loadA(const __bf16* base, int pitch,
                                       int mRow, int k0, int lane) {
  int r  = lane & 15;
  int kb = (lane >> 4) << 3;
  const __bf16* p = base + (mRow + r) * pitch + k0 + kb;
  v8bf lo = *(const v8bf*)p;
  v8bf hi = *(const v8bf*)(p + 16);
  return cat16(lo, hi);
}

// B-fragment (32x16 bf16) from an N-major (transposed) buffer: row n holds K contiguously.
// lane L: col N = L&15, K = kb..kb+7 and 16+kb..16+kb+7, kb = (L>=16)*8.
__device__ __forceinline__ v16bf loadBT(const __bf16* base, int pitch,
                                        int n0, int k0, int lane) {
  int c   = lane & 15;
  int kb2 = (lane >> 4) << 3;
  const __bf16* p = base + (n0 + c) * pitch + k0 + kb2;
  v8bf lo = *(const v8bf*)p;
  v8bf hi = *(const v8bf*)(p + 16);
  return cat16(lo, hi);
}

__device__ __forceinline__ v8f wmma_bf16(v16bf a, v16bf b, v8f c) {
  return __builtin_amdgcn_wmma_f32_16x16x32_bf16(false, a, false, b, (short)0, c, false, false);
}

// D layout: VGPR r -> (M = mRow + (lane>=16?8:0) + r, N = n0 + (lane&15))
// Pairwise v2f->v2bf so each pair is one v_cvt_pk_bf16_f32; hi half stores via d16_hi.
__device__ __forceinline__ void storeD(__bf16* base, int pitch,
                                       int mRow, int n0, v8f c, int lane) {
  int col  = lane & 15;
  int mAdd = (lane >> 4) << 3;
  __bf16* p = base + (mRow + mAdd) * pitch + n0 + col;
#pragma unroll
  for (int r = 0; r < 8; r += 2) {
    v2f t = {c[r], c[r + 1]};
    v2bf pk = __builtin_convertvector(t, v2bf);
    p[r * pitch]       = pk[0];
    p[(r + 1) * pitch] = pk[1];
  }
}

// ---- prologue: transpose f32 [K][N] weight into bf16 [N][K] in workspace ----
__global__ void wtrans_kernel(const float* __restrict__ src, __bf16* __restrict__ dst,
                              int K, int N) {
  int idx = blockIdx.x * blockDim.x + threadIdx.x;
  if (idx >= K * N) return;
  int n = idx / K;
  int k = idx - n * K;
  dst[idx] = (__bf16)src[k * N + n];
}

__global__ __launch_bounds__(256) void gin_wmma_kernel(
    const int* __restrict__ adj, const float* __restrict__ x,
    const __bf16* __restrict__ w1t,   // bf16 [D_HID][D_IN]
    const float* __restrict__ b1,
    const __bf16* __restrict__ w2t,   // bf16 [D_OUT][D_HID]
    const float* __restrict__ b2,
    float* __restrict__ out)
{
  __shared__ __bf16 sS [128 * 136];   // S, row-major (A-src)
  __shared__ __bf16 sT1[128 * 264];   // T1 = S*X row-major (A-src)
  __shared__ __bf16 sU [17920];       // union: XT chunk [128][136] | HT [64][136] + T2 [128][72]
  __shared__ float sB1[D_HID];
  __shared__ float sB2[D_OUT];
  __shared__ int   sMax[D_OUT];

  const int b    = blockIdx.x;
  const int tid  = threadIdx.x;
  const int wave = tid >> 5;
  const int lane = tid & 31;
  const int mRow = wave << 4;          // each wave owns 16 output rows
  const int col  = lane & 15;
  const int mAdd = (lane >> 4) << 3;
  const v8f vzero = {0.f,0.f,0.f,0.f,0.f,0.f,0.f,0.f};

  for (int i = tid; i < D_HID; i += 256) sB1[i] = b1[i];
  if (tid < D_OUT) { sB2[tid] = b2[tid]; sMax[tid] = 0; }

  // Build S[j][i] = (i==j) || (i<j && adj[b][i][j]!=0); int4-vectorized coalesced read.
  const int* adjb = adj + (size_t)b * N_NODES * N_NODES;
  for (int idx = tid; idx < N_NODES * 32; idx += 256) {
    int i  = idx >> 5;
    int j4 = (idx & 31) << 2;
    v4i a4 = *(const v4i*)(adjb + i * N_NODES + j4);
#pragma unroll
    for (int q = 0; q < 4; ++q) {
      int j = j4 + q;
      sS[j * 136 + i] = (i == j || (i < j && a4[q] != 0)) ? (__bf16)1.0f : (__bf16)0.0f;
    }
  }
  __syncthreads();

  // ---- Stage 1: T1 = S * X, feature dim in 2 chunks of 128 ----
  const float* xb = x + (size_t)b * N_NODES * D_IN;
  {
    v16bf aS[4];                       // S fragments: invariant across all of stage 1
#pragma unroll
    for (int k = 0; k < 4; ++k) aS[k] = loadA(sS, 136, mRow, k * 32, lane);

    for (int cx = 0; cx < 2; ++cx) {
      const int d0 = cx * 128;
      for (int idx = tid; idx < N_NODES * 32; idx += 256) {
        int node = idx >> 5;
        int dd4  = (idx & 31) << 2;
        v4f x4 = *(const v4f*)(xb + node * D_IN + d0 + dd4);
        v4bf xh = __builtin_convertvector(x4, v4bf);
#pragma unroll
        for (int q = 0; q < 4; ++q) sU[(dd4 + q) * 136 + node] = xh[q];  // XT chunk
      }
      __syncthreads();
      for (int nt = 0; nt < 8; ++nt) {
        int n0 = nt << 4;
        v16bf bx[4];
#pragma unroll
        for (int k = 0; k < 4; ++k) bx[k] = loadBT(sU, 136, n0, k * 32, lane);
        v8f c = vzero;
#pragma unroll
        for (int k = 0; k < 4; ++k) c = wmma_bf16(aS[k], bx[k], c);
        storeD(sT1, 264, mRow, d0 + n0, c, lane);
      }
      __syncthreads();
    }
  }

  // ---- Hidden dim in 8 chunks of 64; output accumulated in registers ----
  __bf16* sHTc = sU;             // [64][136]  H chunk, transposed (B-src)
  __bf16* sT2c = sU + 64 * 136;  // [128][72]  T2 chunk, row-major (A-src)
  v8f oacc[16];
#pragma unroll
  for (int t = 0; t < 16; ++t) oacc[t] = vzero;

  for (int hc = 0; hc < 8; ++hc) {
    const int h0 = hc << 6;

    // Pull next chunk's weight lines toward the caches while this chunk computes.
    if (hc < 7) {
      const int hn = h0 + 64;
      __builtin_prefetch(w1t + (size_t)hn * D_IN + (lane << 4), 0, 1);
      __builtin_prefetch(w2t + (size_t)(mRow + (lane & 15)) * D_HID + hn, 0, 1);
    }

    // Stage 2: Hc = relu(T1 * W1[:,h0:h0+64] + b1), stored transposed.
    {
      v16bf aT1[8];                    // T1 fragments: invariant across this chunk
#pragma unroll
      for (int k = 0; k < 8; ++k) aT1[k] = loadA(sT1, 264, mRow, k * 32, lane);

      for (int nt = 0; nt < 4; ++nt) {
        const int nG = h0 + (nt << 4);
        v8f c = vzero;
#pragma unroll
        for (int half = 0; half < 2; ++half) {
          v16bf bw[4];
#pragma unroll
          for (int k = 0; k < 4; ++k)
            bw[k] = loadBT(w1t, D_IN, nG, (half * 4 + k) * 32, lane);
#pragma unroll
          for (int k = 0; k < 4; ++k) c = wmma_bf16(aT1[half * 4 + k], bw[k], c);
        }
        const float bias = sB1[nG + col];
#pragma unroll
        for (int r = 0; r < 8; ++r) {
          float v = c[r] + bias;
          c[r] = v > 0.f ? v : 0.f;
        }
        v8bf hv = __builtin_convertvector(c, v8bf);
        *(v8bf*)(sHTc + ((nt << 4) + col) * 136 + (mRow + mAdd)) = hv;  // one b128 store
      }
    }
    __syncthreads();

    // Stage 3: T2c = S * Hc
    {
      v16bf aS[4];                     // S fragments: invariant across this chunk
#pragma unroll
      for (int k = 0; k < 4; ++k) aS[k] = loadA(sS, 136, mRow, k * 32, lane);

      for (int nt = 0; nt < 4; ++nt) {
        const int n0 = nt << 4;
        v16bf bh[4];
#pragma unroll
        for (int k = 0; k < 4; ++k) bh[k] = loadBT(sHTc, 136, n0, k * 32, lane);
        v8f c = vzero;
#pragma unroll
        for (int k = 0; k < 4; ++k) c = wmma_bf16(aS[k], bh[k], c);
        storeD(sT2c, 72, mRow, n0, c, lane);
      }
    }
    __syncthreads();

    // Stage 4: O += T2c * W2[h0:h0+64, :]
    {
      v16bf aT2[2];                    // T2 fragments: invariant across the 16 N-tiles
#pragma unroll
      for (int k = 0; k < 2; ++k) aT2[k] = loadA(sT2c, 72, mRow, k * 32, lane);

      for (int nt = 0; nt < 16; ++nt) {
        const int n0 = nt << 4;
        v16bf bw[2];
#pragma unroll
        for (int k = 0; k < 2; ++k) bw[k] = loadBT(w2t, D_HID, n0, h0 + k * 32, lane);
        v8f c = oacc[nt];
#pragma unroll
        for (int k = 0; k < 2; ++k) c = wmma_bf16(aT2[k], bw[k], c);
        oacc[nt] = c;
      }
    }
    __syncthreads();
  }

  // ---- Epilogue: relu(+b2), per-column max over nodes (nonneg -> int-bit max) ----
  for (int nt = 0; nt < 16; ++nt) {
    const int n = (nt << 4) + col;
    const float bias = sB2[n];
    float m = 0.f;
    v8f c = oacc[nt];
#pragma unroll
    for (int r = 0; r < 8; ++r) {
      float v = c[r] + bias;
      v = v > 0.f ? v : 0.f;
      m = v > m ? v : m;
    }
    atomicMax(&sMax[n], __float_as_int(m));
  }
  __syncthreads();
  if (tid < D_OUT) out[(size_t)b * D_OUT + tid] = __int_as_float(sMax[tid]);
}

extern "C" void kernel_launch(void* const* d_in, const int* in_sizes, int n_in,
                              void* d_out, int out_size, void* d_ws, size_t ws_size,
                              hipStream_t stream) {
  const int*   adj = (const int*)d_in[0];
  const float* x   = (const float*)d_in[1];
  const float* W1  = (const float*)d_in[2];
  const float* b1  = (const float*)d_in[3];
  const float* W2  = (const float*)d_in[4];
  const float* b2  = (const float*)d_in[5];
  float*       out = (float*)d_out;

  __bf16* w1t = (__bf16*)d_ws;                        // [512][256] bf16
  __bf16* w2t = w1t + (size_t)D_HID * D_IN;           // [256][512] bf16

  const int wElems = D_IN * D_HID;                    // 131072 each
  wtrans_kernel<<<(wElems + 255) / 256, 256, 0, stream>>>(W1, w1t, D_IN,  D_HID);
  wtrans_kernel<<<(wElems + 255) / 256, 256, 0, stream>>>(W2, w2t, D_HID, D_OUT);

  gin_wmma_kernel<<<512, 256, 0, stream>>>(adj, x, w1t, b1, w2t, b2, out);
}
